// MultiHeadAttention_61297773248628
// MI455X (gfx1250) — compile-verified, matrix operand order checked by reading, not run on hardware
//
#include <hip/hip_runtime.h>
#include <cstdint>

typedef __bf16 bf16_t;
typedef __attribute__((ext_vector_type(16))) __bf16 v16bf;
typedef __attribute__((ext_vector_type(8)))  float  v8f;

#define B_    32
#define S_    1024
#define H_    512
#define DK_   256
#define BS_   (B_ * S_)
#define NCAT_ 1536          // 6 * DK (q1,k1,v1,q2,k2,v2)
#define EPS_  1e-5f

union BFrag { v16bf v; uint4 q[2]; bf16_t e[16]; };
union Acc   { v8f   v; float f[8]; };

static __device__ __forceinline__ v8f wmma_bf16(v16bf a, v16bf b, v8f c) {
  // (neg_a, A, neg_b, B, c_mod, C, reuse_a, reuse_b) -> v_wmma_f32_16x16x32_bf16
  return __builtin_amdgcn_wmma_f32_16x16x32_bf16(false, a, false, b, (short)0, c, false, false);
}

// CDNA5 async global->LDS copy (16B per lane), tracked by ASYNCcnt.
// lds_off = LDS byte offset (low 32 bits of the shared-aperture flat address).
static __device__ __forceinline__ void async_ld_b128(unsigned lds_off, const void* gaddr) {
  asm volatile("global_load_async_to_lds_b128 %0, %1, off"
               :: "v"(lds_off), "v"((unsigned long long)(uintptr_t)gaddr)
               : "memory");
}

// ---------------------------------------------------------------------------
// Kernel 1: weight prep. WqkvT[n][k] (bf16, n in 0..1535), WoT[n][k] (bf16),
// bcat[1536] f32. B-fragment-friendly [N][K] layout.
// ---------------------------------------------------------------------------
__global__ void prep_kernel(const float* Wq1, const float* Wk1, const float* Wv1,
                            const float* Wq2, const float* Wk2, const float* Wv2,
                            const float* bq1, const float* bk1, const float* bv1,
                            const float* bq2, const float* bk2, const float* bv2,
                            const float* Wo,
                            bf16_t* WqkvT, bf16_t* WoT, float* bcat) {
  const int idx = blockIdx.x * blockDim.x + threadIdx.x;
  const int NW = NCAT_ * H_;     // 786432
  const int NO = H_ * H_;        // 262144
  if (idx < NW) {
    const int n = idx / H_, k = idx % H_;
    const int j = n >> 8, d = n & 255;
    const float* W;
    switch (j) { case 0: W = Wq1; break; case 1: W = Wk1; break; case 2: W = Wv1; break;
                 case 3: W = Wq2; break; case 4: W = Wk2; break; default: W = Wv2; break; }
    WqkvT[(size_t)n * H_ + k] = (bf16_t)W[(size_t)k * DK_ + d];
  } else if (idx < NW + NO) {
    const int i2 = idx - NW;
    const int n = i2 / H_, k = i2 % H_;
    WoT[(size_t)n * H_ + k] = (bf16_t)Wo[(size_t)k * H_ + n];
  } else if (idx < NW + NO + NCAT_) {
    const int n = idx - NW - NO;
    const int j = n >> 8, d = n & 255;
    const float* bb;
    switch (j) { case 0: bb = bq1; break; case 1: bb = bk1; break; case 2: bb = bv1; break;
                 case 3: bb = bq2; break; case 4: bb = bk2; break; default: bb = bv2; break; }
    bcat[n] = bb[d];
  }
}

// ---------------------------------------------------------------------------
// Kernel 2: QKV GEMM. C[BS,1536] = X @ Wqkv + b. 64x64 block tiles, 4 waves,
// K-steps of 32 through LDS. Writes Q,K row-major [S][DK] bf16 and V
// transposed [DK][S] bf16 (per b,head).
// ---------------------------------------------------------------------------
__global__ __launch_bounds__(128) void qkv_gemm_kernel(
    const float* __restrict__ x, const bf16_t* __restrict__ WqkvT,
    const float* __restrict__ bcat,
    bf16_t* __restrict__ Qb, bf16_t* __restrict__ Kb, bf16_t* __restrict__ VTb) {
  __shared__ __align__(16) bf16_t Atile[64][32];
  __shared__ __align__(16) bf16_t Btile[64][32];
  const int t = threadIdx.x;
  const int w = t >> 5, lane = t & 31, l15 = lane & 15, hi = lane >> 4;
  const int M0 = blockIdx.x * 64;
  const int N0 = blockIdx.y * 64;

  Acc acc[4];
#pragma unroll
  for (int ns = 0; ns < 4; ns++)
#pragma unroll
    for (int i = 0; i < 8; i++) acc[ns].f[i] = 0.f;

  const int fr = t >> 1, fc = (t & 1) * 16;

  for (int kk = 0; kk < H_; kk += 32) {
    {   // A: f32 -> bf16 convert on the fly
      const float* src = &x[(size_t)(M0 + fr) * H_ + kk + fc];
#pragma unroll
      for (int i = 0; i < 16; i++) Atile[fr][fc + i] = (bf16_t)src[i];
    }
    {   // B: bf16 copy ([n][k] layout already)
      const uint4* src = (const uint4*)&WqkvT[(size_t)(N0 + fr) * H_ + kk + fc];
      uint4* dst = (uint4*)&Btile[fr][fc];
      dst[0] = src[0]; dst[1] = src[1];
    }
    __syncthreads();
    BFrag a;
    a.q[0] = *(const uint4*)&Atile[w * 16 + l15][8 * hi];
    a.q[1] = *(const uint4*)&Atile[w * 16 + l15][16 + 8 * hi];
#pragma unroll
    for (int ns = 0; ns < 4; ns++) {
      BFrag b;
      b.q[0] = *(const uint4*)&Btile[ns * 16 + l15][8 * hi];
      b.q[1] = *(const uint4*)&Btile[ns * 16 + l15][16 + 8 * hi];
      acc[ns].v = wmma_bf16(a.v, b.v, acc[ns].v);
    }
    __syncthreads();
  }

#pragma unroll
  for (int ns = 0; ns < 4; ns++) {
    const int ng = N0 + ns * 16 + l15;
    const int j = ng >> 8, d = ng & 255;
    const int head = j / 3, type = j % 3;   // 0=q 1=k 2=v
    const float bias = bcat[ng];
    const int srow0 = M0 + w * 16 + 8 * hi; // first of 8 consecutive rows
    if (type == 2) {
      bf16_t tmp[8];
#pragma unroll
      for (int r = 0; r < 8; r++) tmp[r] = (bf16_t)(acc[ns].f[r] + bias);
      const int b = srow0 / S_, s = srow0 % S_;     // 8 rows stay in one batch
      bf16_t* dstp = &VTb[(((size_t)(b * 2 + head)) * DK_ + d) * S_ + s];
      *(uint4*)dstp = *(uint4*)tmp;                 // contiguous along s
    } else {
      bf16_t* base = (type == 0) ? Qb : Kb;
#pragma unroll
      for (int r = 0; r < 8; r++) {
        const int sg = srow0 + r;
        const int b = sg / S_, s = sg % S_;
        base[(((size_t)(b * 2 + head)) * S_ + s) * DK_ + d] = (bf16_t)(acc[ns].f[r] + bias);
      }
    }
  }
}

// ---------------------------------------------------------------------------
// Kernel 3: flash attention. 256 threads = 8 waves share one (b,head); each
// wave owns 16 Q rows. Per 32-wide K/V column pair: K-tile (32x256) and
// V^T-tile (256x32) are staged into LDS with async global->LDS b128 copies
// shared by all 8 waves (8x less L2 traffic), then online softmax + WMMA.
// ---------------------------------------------------------------------------
__global__ __launch_bounds__(256) void attn_kernel(
    const bf16_t* __restrict__ Qb, const bf16_t* __restrict__ Kb,
    const bf16_t* __restrict__ VTb, bf16_t* __restrict__ Cc) {
  __shared__ __align__(16) bf16_t Kt[32][256];    // 16 KB  [t][d]
  __shared__ __align__(16) bf16_t Vt[256][32];    // 16 KB  [d][t]
  __shared__ __align__(16) bf16_t Pt[8][16][32];  //  8 KB  per-wave P bounce
  const int t = threadIdx.x;
  const int w = t >> 5, lane = t & 31, l15 = lane & 15, hi = lane >> 4;
  const int bh = blockIdx.x >> 3;       // b*2 + head
  const int sblk = blockIdx.x & 7;
  const int s0 = sblk * 128 + w * 16;

  const bf16_t* Qp = Qb  + (size_t)bh * S_ * DK_;
  const bf16_t* Kp = Kb  + (size_t)bh * S_ * DK_;
  const bf16_t* Vp = VTb + (size_t)bh * DK_ * S_;

  BFrag qf[8];                          // Q row block, all of DK
#pragma unroll
  for (int d8 = 0; d8 < 8; d8++) {
    const bf16_t* p = &Qp[(size_t)(s0 + l15) * DK_ + 32 * d8 + 8 * hi];
    qf[d8].q[0] = *(const uint4*)p;
    qf[d8].q[1] = *(const uint4*)(p + 16);
  }

  Acc acc[16];                          // O: 16 rows x 256 cols f32
#pragma unroll
  for (int n = 0; n < 16; n++)
#pragma unroll
    for (int i = 0; i < 8; i++) acc[n].f[i] = 0.f;

  float mrun[8], lrun[8];
#pragma unroll
  for (int r = 0; r < 8; r++) { mrun[r] = -1e30f; lrun[r] = 0.f; }

  // Staging geometry: K: thread -> row t>>3 (0..31), 64B chunk (t&7)*64.
  //                   V: thread -> row t (0..255), full 64B row.
  const int krow = t >> 3, kcb = (t & 7) * 32;           // elements
  const unsigned kldsbase = (unsigned)(uintptr_t)&Kt[krow][kcb];
  const unsigned vldsbase = (unsigned)(uintptr_t)&Vt[t][0];

  for (int t2 = 0; t2 < S_; t2 += 32) {
    {   // async stage K tile rows t2..t2+31  (4 x b128 per thread)
      const bf16_t* g = &Kp[(size_t)(t2 + krow) * DK_ + kcb];
#pragma unroll
      for (int c = 0; c < 4; c++) async_ld_b128(kldsbase + c * 16, g + c * 8);
    }
    {   // async stage V^T tile cols t2..t2+31 (4 x b128 per thread)
      const bf16_t* g = &Vp[(size_t)t * S_ + t2];
#pragma unroll
      for (int c = 0; c < 4; c++) async_ld_b128(vldsbase + c * 16, g + c * 8);
    }
    asm volatile("s_wait_asynccnt 0" ::: "memory");
    __syncthreads();

    Acc sc0, sc1;
#pragma unroll
    for (int i = 0; i < 8; i++) { sc0.f[i] = 0.f; sc1.f[i] = 0.f; }
#pragma unroll
    for (int d8 = 0; d8 < 8; d8++) {    // scores tile t2..t2+15
      BFrag kf;
      const bf16_t* p = &Kt[l15][32 * d8 + 8 * hi];
      kf.q[0] = *(const uint4*)p; kf.q[1] = *(const uint4*)(p + 16);
      sc0.v = wmma_bf16(qf[d8].v, kf.v, sc0.v);
    }
#pragma unroll
    for (int d8 = 0; d8 < 8; d8++) {    // scores tile t2+16..t2+31
      BFrag kf;
      const bf16_t* p = &Kt[16 + l15][32 * d8 + 8 * hi];
      kf.q[0] = *(const uint4*)p; kf.q[1] = *(const uint4*)(p + 16);
      sc1.v = wmma_bf16(qf[d8].v, kf.v, sc1.v);
    }

    float alpha[8];
#pragma unroll
    for (int r = 0; r < 8; r++) {       // online softmax (row m = r + 8*hi)
      float a = sc0.f[r] * 0.0625f, b = sc1.f[r] * 0.0625f;   // 1/sqrt(256)
      float v = fmaxf(a, b);
#pragma unroll
      for (int off = 1; off < 16; off <<= 1) v = fmaxf(v, __shfl_xor(v, off, 32));
      const float mn = fmaxf(mrun[r], v);
      alpha[r] = __expf(mrun[r] - mn);
      const float p0 = __expf(a - mn), p1 = __expf(b - mn);
      sc0.f[r] = p0; sc1.f[r] = p1;
      float rs = p0 + p1;
#pragma unroll
      for (int off = 1; off < 16; off <<= 1) rs += __shfl_xor(rs, off, 32);
      lrun[r] = lrun[r] * alpha[r] + rs;
      mrun[r] = mn;
    }
#pragma unroll
    for (int n = 0; n < 16; n++)        // rescale O accumulators
#pragma unroll
      for (int r = 0; r < 8; r++) acc[n].f[r] *= alpha[r];

#pragma unroll
    for (int r = 0; r < 8; r++) {       // P: C layout -> LDS [m][k]
      Pt[w][r + 8 * hi][l15]      = (bf16_t)sc0.f[r];
      Pt[w][r + 8 * hi][16 + l15] = (bf16_t)sc1.f[r];
    }
    // same-wave LDS is in-order; barrier only stops compiler reordering
    asm volatile("s_wait_dscnt 0" ::: "memory");

    BFrag pf;                           // P as A-fragment (16x32)
    pf.q[0] = *(const uint4*)&Pt[w][l15][8 * hi];
    pf.q[1] = *(const uint4*)&Pt[w][l15][16 + 8 * hi];

#pragma unroll
    for (int n = 0; n < 16; n++) {      // O += P x V  (V^T rows contiguous)
      BFrag vf;
      const bf16_t* p = &Vt[n * 16 + l15][8 * hi];
      vf.q[0] = *(const uint4*)p; vf.q[1] = *(const uint4*)(p + 16);
      acc[n].v = wmma_bf16(pf.v, vf.v, acc[n].v);
    }
    __syncthreads();                    // before re-staging K/V tiles
  }

  const int b = bh >> 1, head = bh & 1;
  float rinv[8];
#pragma unroll
  for (int r = 0; r < 8; r++) rinv[r] = 1.0f / lrun[r];
  const size_t rowbase = (size_t)b * S_ + s0;
#pragma unroll
  for (int n = 0; n < 16; n++) {
    const int col = head * DK_ + n * 16 + l15;
#pragma unroll
    for (int r = 0; r < 8; r++) {
      const int m = r + 8 * hi;
      Cc[(rowbase + m) * (size_t)(2 * DK_) + col] = (bf16_t)(acc[n].f[r] * rinv[r]);
    }
  }
}

// ---------------------------------------------------------------------------
// Kernel 4a: output projection + bias + residual -> f32 d_out
// ---------------------------------------------------------------------------
__global__ __launch_bounds__(128) void proj_gemm_kernel(
    const bf16_t* __restrict__ Cc, const bf16_t* __restrict__ WoT,
    const float* __restrict__ bo, const float* __restrict__ x,
    float* __restrict__ out) {
  __shared__ __align__(16) bf16_t Atile[64][32];
  __shared__ __align__(16) bf16_t Btile[64][32];
  const int t = threadIdx.x;
  const int w = t >> 5, lane = t & 31, l15 = lane & 15, hi = lane >> 4;
  const int M0 = blockIdx.x * 64;
  const int N0 = blockIdx.y * 64;

  Acc acc[4];
#pragma unroll
  for (int ns = 0; ns < 4; ns++)
#pragma unroll
    for (int i = 0; i < 8; i++) acc[ns].f[i] = 0.f;

  const int fr = t >> 1, fc = (t & 1) * 16;

  for (int kk = 0; kk < 2 * DK_; kk += 32) {
    { const uint4* s1 = (const uint4*)&Cc[(size_t)(M0 + fr) * (2 * DK_) + kk + fc];
      uint4* d1 = (uint4*)&Atile[fr][fc]; d1[0] = s1[0]; d1[1] = s1[1]; }
    { const uint4* s2 = (const uint4*)&WoT[(size_t)(N0 + fr) * (2 * DK_) + kk + fc];
      uint4* d2 = (uint4*)&Btile[fr][fc]; d2[0] = s2[0]; d2[1] = s2[1]; }
    __syncthreads();
    BFrag a;
    a.q[0] = *(const uint4*)&Atile[w * 16 + l15][8 * hi];
    a.q[1] = *(const uint4*)&Atile[w * 16 + l15][16 + 8 * hi];
#pragma unroll
    for (int ns = 0; ns < 4; ns++) {
      BFrag b;
      b.q[0] = *(const uint4*)&Btile[ns * 16 + l15][8 * hi];
      b.q[1] = *(const uint4*)&Btile[ns * 16 + l15][16 + 8 * hi];
      acc[ns].v = wmma_bf16(a.v, b.v, acc[ns].v);
    }
    __syncthreads();
  }

#pragma unroll
  for (int ns = 0; ns < 4; ns++) {
    const int ng = N0 + ns * 16 + l15;
    const float bias = bo[ng];
    const int row0 = M0 + w * 16 + 8 * hi;
#pragma unroll
    for (int r = 0; r < 8; r++) {
      const size_t g = (size_t)(row0 + r) * H_ + ng;
      out[g] = acc[ns].f[r] + bias + x[g];
    }
  }
}

// ---------------------------------------------------------------------------
// Kernel 4b: layernorm in-place on d_out; one wave per 512-wide row
// ---------------------------------------------------------------------------
__global__ __launch_bounds__(128) void layernorm_kernel(
    float* __restrict__ out, const float* __restrict__ gamma,
    const float* __restrict__ beta) {
  const int t = threadIdx.x;
  const int w = t >> 5, lane = t & 31;
  const size_t row = (size_t)blockIdx.x * 4 + w;
  float* p = out + row * H_;
  float v[16];
  float s = 0.f, ss = 0.f;
#pragma unroll
  for (int i = 0; i < 16; i++) {
    v[i] = p[lane * 16 + i];
    s += v[i]; ss += v[i] * v[i];
  }
#pragma unroll
  for (int off = 1; off < 32; off <<= 1) {
    s  += __shfl_xor(s, off, 32);
    ss += __shfl_xor(ss, off, 32);
  }
  const float mu   = s  * (1.0f / H_);
  const float var  = ss * (1.0f / H_) - mu * mu;
  const float rstd = rsqrtf(var + EPS_);
#pragma unroll
  for (int i = 0; i < 16; i++) {
    const int c = lane * 16 + i;
    p[c] = gamma[c] * (v[i] - mu) * rstd + beta[c];
  }
}

// ---------------------------------------------------------------------------
extern "C" void kernel_launch(void* const* d_in, const int* in_sizes, int n_in,
                              void* d_out, int out_size, void* d_ws, size_t ws_size,
                              hipStream_t stream) {
  (void)in_sizes; (void)n_in; (void)out_size; (void)ws_size;
  const float* x   = (const float*)d_in[0];
  const float* Wq1 = (const float*)d_in[1];  const float* bq1 = (const float*)d_in[2];
  const float* Wk1 = (const float*)d_in[3];  const float* bk1 = (const float*)d_in[4];
  const float* Wv1 = (const float*)d_in[5];  const float* bv1 = (const float*)d_in[6];
  const float* Wq2 = (const float*)d_in[7];  const float* bq2 = (const float*)d_in[8];
  const float* Wk2 = (const float*)d_in[9];  const float* bk2 = (const float*)d_in[10];
  const float* Wv2 = (const float*)d_in[11]; const float* bv2 = (const float*)d_in[12];
  const float* Wo  = (const float*)d_in[13]; const float* bo  = (const float*)d_in[14];
  const float* gamma = (const float*)d_in[15];
  const float* beta  = (const float*)d_in[16];
  float* out = (float*)d_out;

  uint8_t* ws = (uint8_t*)d_ws;
  size_t off = 0;
  auto alloc = [&](size_t bytes) -> void* {
    void* p = ws + off;
    off = (off + bytes + 255) & ~(size_t)255;
    return p;
  };
  bf16_t* WqkvT = (bf16_t*)alloc((size_t)NCAT_ * H_ * 2);        // 1.5 MB
  bf16_t* WoT   = (bf16_t*)alloc((size_t)H_ * H_ * 2);           // 0.5 MB
  float*  bcat  = (float*) alloc((size_t)NCAT_ * 4);
  bf16_t* Qb    = (bf16_t*)alloc((size_t)B_ * 2 * S_ * DK_ * 2); // 32 MB
  bf16_t* Kb    = (bf16_t*)alloc((size_t)B_ * 2 * S_ * DK_ * 2); // 32 MB
  bf16_t* VTb   = (bf16_t*)alloc((size_t)B_ * 2 * DK_ * S_ * 2); // 32 MB (V^T)
  bf16_t* Cc    = (bf16_t*)alloc((size_t)BS_ * 2 * DK_ * 2);     // 32 MB (concat)

  {
    const int total = NCAT_ * H_ + H_ * H_ + NCAT_;
    prep_kernel<<<(total + 255) / 256, 256, 0, stream>>>(
        Wq1, Wk1, Wv1, Wq2, Wk2, Wv2,
        bq1, bk1, bv1, bq2, bk2, bv2, Wo, WqkvT, WoT, bcat);
  }
  {
    dim3 g(BS_ / 64, NCAT_ / 64);
    qkv_gemm_kernel<<<g, 128, 0, stream>>>(x, WqkvT, bcat, Qb, Kb, VTb);
  }
  {
    attn_kernel<<<B_ * 2 * (S_ / 128), 256, 0, stream>>>(Qb, Kb, VTb, Cc);
  }
  {
    dim3 g(BS_ / 64, H_ / 64);
    proj_gemm_kernel<<<g, 128, 0, stream>>>(Cc, WoT, bo, x, out);
  }
  {
    layernorm_kernel<<<BS_ / 4, 128, 0, stream>>>(out, gamma, beta);
  }
}
